// RuleMiningAgent_154618823006
// MI455X (gfx1250) — compile-verified, matrix operand order checked by reading, not run on hardware
//
#include <hip/hip_runtime.h>

// MI455X / gfx1250, wave32. Matrix work runs on v_wmma_f32_16x16x32_bf16.
//
// Pipeline (all in d_ws, f32->bf16 once):
//   1) Xc = bf16(concat(H, rel_emb[q]))            [4096 x 512]
//   2) X  = relu(Xc @ W1^T + b1)   (bf16 WMMA)     [4096 x 256]
//   3) X2 = X @ W2^T + b2          (bf16 WMMA)     [4096 x 256]
//   4) S  = X2 @ rel_emb^T         (bf16 WMMA)     [4096 x 1216] (N padded 1200->1216; dense GEMM
//                                                   replaces gathered matvec; S stays in 192MB L2)
//   5) scores[b,k] = S[b, r_space[b,k]] - (1-mask)*1e31 -> softmax -> dist + entropy
//
// GEMM register blocking: each wave owns a 32M x 32N tile (2x2 grid of 16x16 WMMA tiles),
// so each A/B fragment pair feeds 2 WMMAs -> 8 b128 loads per 4 WMMAs per K-step.

typedef __attribute__((ext_vector_type(16))) __bf16 v16bf;
typedef __attribute__((ext_vector_type(8)))  __bf16 v8bf;
typedef __attribute__((ext_vector_type(8)))  float  v8f;

static __device__ __forceinline__ unsigned short f2bf_bits(float f) {
  unsigned u = __builtin_bit_cast(unsigned, f);
  u += 0x7FFFu + ((u >> 16) & 1u);          // round-to-nearest-even
  return (unsigned short)(u >> 16);
}

__global__ __launch_bounds__(256) void cvt_f32_bf16(const float* __restrict__ in,
                                                    unsigned short* __restrict__ out, int n) {
  int i = blockIdx.x * 256 + threadIdx.x;
  if (i < n) out[i] = f2bf_bits(in[i]);
}

// Xc[b, 0:256] = H[b, :]; Xc[b, 256:512] = rel_emb[q[b], :]
__global__ __launch_bounds__(256) void build_xc(const float* __restrict__ H,
                                                const int* __restrict__ q,
                                                const float* __restrict__ rel,
                                                unsigned short* __restrict__ xc) {
  int i = blockIdx.x * 256 + threadIdx.x;   // over 4096*512
  int b = i >> 9, j = i & 511;
  float v = (j < 256) ? H[(b << 8) + j] : rel[q[b] * 256 + (j - 256)];
  xc[i] = f2bf_bits(v);
}

// Load a 16x32 bf16 fragment per ISA 7.12.2: lane L -> row L%16, two contiguous
// 16B chunks at K = (L/16)*8 and (L/16)*8 + 16. Same pattern serves A (row-major
// [M,Kd]) and B (row-major [N,Kd], since the ISA B layout is the mirror of A).
static __device__ __forceinline__ v16bf load_frag(const unsigned short* p) {
  v8bf lo = *(const v8bf*)(p);
  v8bf hi = *(const v8bf*)(p + 16);
  v16bf f;
#pragma unroll
  for (int i = 0; i < 8; ++i) { f[i] = lo[i]; f[i + 8] = hi[i]; }
  return f;
}

// C[M,N] = act(A[M,Kd] * B[N,Kd]^T + bias[N]); A,B row-major bf16, Kd = KDIM.
// One wave computes a 32x32 tile (2x2 WMMA sub-tiles); fully unrolled K loop.
template <int KDIM, bool RELU, bool HAS_BIAS, bool OUT_BF16>
__global__ __launch_bounds__(256) void gemm_bf16_wmma(const unsigned short* __restrict__ A,
                                                      const unsigned short* __restrict__ Bm,
                                                      const float* __restrict__ bias,
                                                      void* __restrict__ outp, int M, int N) {
  const int wave = threadIdx.x >> 5;
  const int lane = threadIdx.x & 31;
  const int tile = blockIdx.x * 8 + wave;
  const int tilesN = N >> 5;                 // 32-wide N blocks
  const int tm = tile / tilesN;
  const int tn = tile - tm * tilesN;
  if (tm * 32 >= M) return;                  // wave-uniform (EXEC stays all-1s for WMMA)

  const int r    = lane & 15;                // A: row-in-tile; B: col-in-tile; C: N index
  const int half = lane >> 4;                // K-phase per ISA 16-bit fragment layout
  const unsigned short* arow0 = A  + (size_t)(tm * 32 + r)      * KDIM + half * 8;
  const unsigned short* arow1 = A  + (size_t)(tm * 32 + 16 + r) * KDIM + half * 8;
  const unsigned short* brow0 = Bm + (size_t)(tn * 32 + r)      * KDIM + half * 8;
  const unsigned short* brow1 = Bm + (size_t)(tn * 32 + 16 + r) * KDIM + half * 8;

  v8f c00, c01, c10, c11;
  const float b0 = HAS_BIAS ? bias[tn * 32 + r] : 0.0f;       // bias is per-N
  const float b1 = HAS_BIAS ? bias[tn * 32 + 16 + r] : 0.0f;
#pragma unroll
  for (int v = 0; v < 8; ++v) { c00[v] = b0; c01[v] = b1; c10[v] = b0; c11[v] = b1; }

#pragma unroll
  for (int k = 0; k < KDIM; k += 32) {
    v16bf a0 = load_frag(arow0 + k);
    v16bf a1 = load_frag(arow1 + k);
    v16bf bb0 = load_frag(brow0 + k);
    v16bf bb1 = load_frag(brow1 + k);
    c00 = __builtin_amdgcn_wmma_f32_16x16x32_bf16(false, a0, false, bb0, (short)0, c00, false, false);
    c01 = __builtin_amdgcn_wmma_f32_16x16x32_bf16(false, a0, false, bb1, (short)0, c01, false, false);
    c10 = __builtin_amdgcn_wmma_f32_16x16x32_bf16(false, a1, false, bb0, (short)0, c10, false, false);
    c11 = __builtin_amdgcn_wmma_f32_16x16x32_bf16(false, a1, false, bb1, (short)0, c11, false, false);
  }

  // C layout: lane L, VGPR v -> (M = v + 8*(L/16), N = L%16) within each 16x16 sub-tile.
  const int n0 = tn * 32 + r;
  const int m0 = tm * 32 + half * 8;
#pragma unroll
  for (int v = 0; v < 8; ++v) {
    float v00 = c00[v], v01 = c01[v], v10 = c10[v], v11 = c11[v];
    if (RELU) {
      v00 = fmaxf(v00, 0.0f); v01 = fmaxf(v01, 0.0f);
      v10 = fmaxf(v10, 0.0f); v11 = fmaxf(v11, 0.0f);
    }
    size_t o00 = (size_t)(m0 + v) * N + n0;
    size_t o10 = (size_t)(m0 + 16 + v) * N + n0;
    if (OUT_BF16) {
      unsigned short* o = (unsigned short*)outp;
      o[o00] = f2bf_bits(v00); o[o00 + 16] = f2bf_bits(v01);
      o[o10] = f2bf_bits(v10); o[o10 + 16] = f2bf_bits(v11);
    } else {
      float* o = (float*)outp;
      o[o00] = v00; o[o00 + 16] = v01;
      o[o10] = v10; o[o10 + 16] = v11;
    }
  }
}

// One block per batch row: gather scores from S_all (ld = NP), mask, softmax, entropy.
__global__ __launch_bounds__(256) void gather_softmax_entropy(const float* __restrict__ S,
                                                              const int* __restrict__ r_space,
                                                              const float* __restrict__ mask,
                                                              float* __restrict__ dist,
                                                              float* __restrict__ ent, int NP) {
  __shared__ float red[256];
  const int b = blockIdx.x, k = threadIdx.x;
  const int rid = r_space[b * 256 + k];
  float s = S[(size_t)b * NP + rid];
  s -= (1.0f - mask[b * 256 + k]) * 1e31f;

  red[k] = s; __syncthreads();
  for (int o = 128; o > 0; o >>= 1) { if (k < o) red[k] = fmaxf(red[k], red[k + o]); __syncthreads(); }
  const float mx = red[0]; __syncthreads();

  const float e = expf(s - mx);
  red[k] = e; __syncthreads();
  for (int o = 128; o > 0; o >>= 1) { if (k < o) red[k] += red[k + o]; __syncthreads(); }
  const float inv = 1.0f / red[0]; __syncthreads();

  const float p = e * inv;
  dist[b * 256 + k] = p;
  red[k] = -p * logf(p + 1e-20f); __syncthreads();
  for (int o = 128; o > 0; o >>= 1) { if (k < o) red[k] += red[k + o]; __syncthreads(); }
  if (k == 0) ent[b] = red[0];
}

extern "C" void kernel_launch(void* const* d_in, const int* in_sizes, int n_in,
                              void* d_out, int out_size, void* d_ws, size_t ws_size,
                              hipStream_t stream) {
  const float* H    = (const float*)d_in[0];
  const int*   q    = (const int*)d_in[1];
  const int*   rsp  = (const int*)d_in[2];
  const float* amsk = (const float*)d_in[3];
  const float* rel  = (const float*)d_in[4];
  const float* W1w  = (const float*)d_in[5];
  const float* W1b  = (const float*)d_in[6];
  const float* W2w  = (const float*)d_in[7];
  const float* W2b  = (const float*)d_in[8];

  constexpr int B = 4096, K = 256, D = 256, NR = 1200;
  constexpr int NP = 1216;                   // NR padded to a multiple of 32 for GEMM3

  // Workspace carve-up (~30 MB total), 256B aligned slices.
  char* ws = (char*)d_ws;
  size_t off = 0;
  auto carve = [&](size_t bytes) -> void* {
    void* p = ws + off;
    off = (off + bytes + 255) & ~(size_t)255;
    return p;
  };
  unsigned short* xc    = (unsigned short*)carve((size_t)B * 2 * D * 2);   // 4 MB
  unsigned short* w1bf  = (unsigned short*)carve((size_t)D * 2 * D * 2);   // 256 KB
  unsigned short* w2bf  = (unsigned short*)carve((size_t)D * D * 2);       // 128 KB
  unsigned short* relbf = (unsigned short*)carve((size_t)NP * D * 2);      // 608 KB (padded)
  unsigned short* xbf   = (unsigned short*)carve((size_t)B * D * 2);       // 2 MB
  unsigned short* x2bf  = (unsigned short*)carve((size_t)B * D * 2);       // 2 MB
  float*          sall  = (float*)carve((size_t)B * NP * sizeof(float));   // 19.9 MB

  // 1) one-time f32 -> bf16 conversions (+ zero the 16 padded rel rows)
  cvt_f32_bf16<<<(D * 2 * D + 255) / 256, 256, 0, stream>>>(W1w, w1bf, D * 2 * D);
  cvt_f32_bf16<<<(D * D + 255) / 256, 256, 0, stream>>>(W2w, w2bf, D * D);
  cvt_f32_bf16<<<(NR * D + 255) / 256, 256, 0, stream>>>(rel, relbf, NR * D);
  hipMemsetAsync(relbf + (size_t)NR * D, 0, (size_t)(NP - NR) * D * 2, stream);
  build_xc<<<(B * 2 * D) / 256, 256, 0, stream>>>(H, q, rel, xc);

  // 2) X = relu(Xc @ W1^T + b1): (4096/32)*(256/32) = 1024 wave-tiles -> 128 blocks
  gemm_bf16_wmma<2 * D, true, true, true>
      <<<(B / 32) * (D / 32) / 8, 256, 0, stream>>>(xc, w1bf, W1b, xbf, B, D);
  // 3) X2 = X @ W2^T + b2
  gemm_bf16_wmma<D, false, true, true>
      <<<(B / 32) * (D / 32) / 8, 256, 0, stream>>>(xbf, w2bf, W2b, x2bf, B, D);
  // 4) S_all = X2 @ rel^T : (4096/32)*(1216/32) = 4864 wave-tiles -> 608 blocks
  gemm_bf16_wmma<D, false, false, false>
      <<<(B / 32) * (NP / 32) / 8, 256, 0, stream>>>(x2bf, relbf, nullptr, sall, B, NP);

  // 5) gather + mask + softmax + entropy; out = [dist (B*K) | entropy (B)]
  gather_softmax_entropy<<<B, 256, 0, stream>>>(sall, rsp, amsk,
                                                (float*)d_out,
                                                (float*)d_out + (size_t)B * K, NP);
}
